// RNNLayer_61375082660034
// MI455X (gfx1250) — matrix operand order repaired and checked by reading, compiled, not run on hardware
//
#include <hip/hip_runtime.h>
#include <hip/hip_bf16.h>

// ---------------------------------------------------------------------------
// 2-layer tanh RNN for MI455X (gfx1250), wave32 + v_wmma_f32_16x16x32_bf16.
//
// Phase 1 (xp_gemm): xp[t,b,h] = x[b,t,:]·W_ih0[h,:] + b_ih0[h] + b_hh0[h]
//   - one block per timestep, 16 waves = (2 batch-half tiles) x (8 H tiles)
//   - x[t] staged to LDS as bf16, W_ih0 read as bf16 from ws (L2-resident)
//   - result stored PRE-SWIZZLED in the WMMA C/D lane layout so the scan
//     kernel can load its accumulator init with contiguous b128 loads.
// Phase 2 (rnn_scan): persistent kernel, 2 blocks x 8 waves, 16 batch rows
//   per block. Recurrent weights live in VGPRs (v16bf x 12 per wave),
//   h-state double-buffered in LDS (bank-conflict-free padded rows).
//   Critical path per step: the h1(t-1)@Whh1^T chain is hoisted BEFORE
//   barrier1 (it only needs last step's state), so after barrier1 only the
//   h0n@Wih1^T chain + tanh remain. xp is register-prefetched 1 step ahead
//   and L2-prefetched (global_prefetch_b8) 8 steps ahead.
// ---------------------------------------------------------------------------

typedef __bf16 bf16_t;
typedef __bf16 v8bf  __attribute__((ext_vector_type(8)));
typedef __bf16 v16bf __attribute__((ext_vector_type(16)));
typedef float  v4f   __attribute__((ext_vector_type(4)));
typedef float  v8f   __attribute__((ext_vector_type(8)));

#define RNN_B 32
#define RNN_T 2048
#define RNN_D 256
#define RNN_H 128

// ws layout (bytes)
#define XP_OFF   0u                       // T*B*H fp32, pre-swizzled (32 MB)
#define WIH0_OFF 33554432u                // 128*256 bf16
#define WHH0_OFF (WIH0_OFF + 65536u)      // 128*128 bf16
#define WIH1_OFF (WHH0_OFF + 32768u)
#define WHH1_OFF (WIH1_OFF + 32768u)
#define BS0_OFF  (WHH1_OFF + 32768u)      // 128 fp32 (b_ih0 + b_hh0)
#define BS1_OFF  (BS0_OFF + 512u)         // 128 fp32 (b_ih1 + b_hh1)

#define WMMA_BF16(a, b, c) \
  __builtin_amdgcn_wmma_f32_16x16x32_bf16(false, (a), false, (b), (short)0, (c), false, false)

__device__ __forceinline__ v16bf cat8(v8bf lo, v8bf hi) {
  return __builtin_shufflevector(lo, hi, 0, 1, 2, 3, 4, 5, 6, 7,
                                 8, 9, 10, 11, 12, 13, 14, 15);
}

// A operand (16x32 bf16) from an LDS tile with padded row stride (elements).
// Layout (matches load_B_g): lanes 0-15: M=lane, K = k0+{0..15}
//                        lanes 16-31: M=lane-16, K = k0+{16..31}
__device__ __forceinline__ v16bf load_A_lds(const bf16_t* base, int lane,
                                            int k0, int stride) {
  int m  = lane & 15;
  int c1 = k0 + ((lane < 16) ? 0 : 16);
  const v8bf* p1 = (const v8bf*)(base + m * stride + c1);
  const v8bf* p2 = (const v8bf*)(base + m * stride + c1 + 8);
  return cat8(*p1, *p2);
}

// B operand (32x16 bf16) = W^T tile; W is [N][K] row-major bf16 in global.
// Lanes 0-15: column N=nbase+lane, K=k0..k0+15; lanes 16-31: K=k0+16..k0+31.
__device__ __forceinline__ v16bf load_B_g(const bf16_t* __restrict__ W, int K,
                                          int nbase, int lane, int k0) {
  int n = nbase + (lane & 15);
  int c = k0 + ((lane < 16) ? 0 : 16);
  return *(const v16bf*)(W + (size_t)n * K + c);
}

__device__ __forceinline__ float fast_tanh(float x) {
  // tanh(x) = 1 - 2/(exp(2x)+1); exact at both saturation ends.
  float e = __expf(2.0f * x);
  return 1.0f - 2.0f * __builtin_amdgcn_rcpf(e + 1.0f);
}

// ---------------------------------------------------------------------------
// Prep: fp32 -> bf16 weight conversion + bias sums.
// ---------------------------------------------------------------------------
__global__ __launch_bounds__(256) void rnn_prep(
    const float* __restrict__ Wih0f, const float* __restrict__ Whh0f,
    const float* __restrict__ bih0, const float* __restrict__ bhh0,
    const float* __restrict__ Wih1f, const float* __restrict__ Whh1f,
    const float* __restrict__ bih1, const float* __restrict__ bhh1,
    bf16_t* __restrict__ wih0, bf16_t* __restrict__ whh0,
    bf16_t* __restrict__ wih1, bf16_t* __restrict__ whh1,
    float* __restrict__ bs0, float* __restrict__ bs1) {
  int i = blockIdx.x * 256 + threadIdx.x;
  if (i < 32768) {
    wih0[i] = (bf16_t)Wih0f[i];
  } else if (i < 49152) {
    int j = i - 32768; whh0[j] = (bf16_t)Whh0f[j];
  } else if (i < 65536) {
    int j = i - 49152; wih1[j] = (bf16_t)Wih1f[j];
  } else if (i < 81920) {
    int j = i - 65536; whh1[j] = (bf16_t)Whh1f[j];
  } else if (i < 82048) {
    int h = i - 81920; bs0[h] = bih0[h] + bhh0[h];
  } else if (i < 82176) {
    int h = i - 82048; bs1[h] = bih1[h] + bhh1[h];
  }
}

// ---------------------------------------------------------------------------
// Phase 1: xp[t] = x[:,t,:] @ W_ih0^T + (b_ih0+b_hh0), stored pre-swizzled:
//   xp[((t*16 + tile)*32 + lane)*8 + r]   tile = mi*8 + ni
// ---------------------------------------------------------------------------
__global__ __launch_bounds__(512) void xp_gemm(
    const float* __restrict__ x, const bf16_t* __restrict__ Wih0,
    const float* __restrict__ bs0, float* __restrict__ xp) {
  __shared__ bf16_t xs[RNN_B][RNN_D + 8];  // padded: 264*2B row = 33x16B

  const int t   = blockIdx.x;
  const int tid = threadIdx.x;

  // Stage x[:, t, :] -> LDS as bf16 (each thread: one 16-float chunk).
  {
    int row = tid >> 4;          // 0..31 (batch)
    int col = (tid & 15) * 16;   // 0..240
    const float* src = x + ((size_t)row * RNN_T + t) * RNN_D + col;
    v4f f0 = *(const v4f*)(src + 0);
    v4f f1 = *(const v4f*)(src + 4);
    v4f f2 = *(const v4f*)(src + 8);
    v4f f3 = *(const v4f*)(src + 12);
    bf16_t* dst = &xs[row][col];
#pragma unroll
    for (int j = 0; j < 4; ++j) {
      dst[j + 0]  = (bf16_t)f0[j];
      dst[j + 4]  = (bf16_t)f1[j];
      dst[j + 8]  = (bf16_t)f2[j];
      dst[j + 12] = (bf16_t)f3[j];
    }
  }
  __syncthreads();

  const int wave = tid >> 5;     // 0..15
  const int lane = tid & 31;
  const int mi   = wave >> 3;    // batch half
  const int ni   = wave & 7;     // H tile
  const int ncol = ni * 16 + (lane & 15);

  float bs = bs0[ncol];
  v8f acc = {bs, bs, bs, bs, bs, bs, bs, bs};

#pragma unroll
  for (int ks = 0; ks < 8; ++ks) {
    int k0 = ks * 32;
    v16bf a = load_A_lds(&xs[mi * 16][0], lane, k0, RNN_D + 8);
    v16bf b = load_B_g(Wih0, RNN_D, ni * 16, lane, k0);
    acc = WMMA_BF16(a, b, acc);
  }

  float* dst = xp + (((size_t)t * 16 + wave) * 32 + lane) * 8;
  v4f lo = {acc[0], acc[1], acc[2], acc[3]};
  v4f hi = {acc[4], acc[5], acc[6], acc[7]};
  *(v4f*)(dst + 0) = lo;
  *(v4f*)(dst + 4) = hi;
}

// ---------------------------------------------------------------------------
// Phase 2: persistent recurrent scan. grid=2 (batch halves), block=256 (8 waves,
// one per 16-wide H tile). Recurrent weights in VGPRs; state in LDS.
// ---------------------------------------------------------------------------
#define HS 136  // padded LDS row stride (bf16 elems): 272B = 17x16B, no conflicts

__global__ __launch_bounds__(256) void rnn_scan(
    const float* __restrict__ xp, const bf16_t* __restrict__ Whh0,
    const bf16_t* __restrict__ Wih1, const bf16_t* __restrict__ Whh1,
    const float* __restrict__ bs1, float* __restrict__ out) {
  __shared__ bf16_t h0s[2][16][HS];
  __shared__ bf16_t h1s[2][16][HS];

  const int tid  = threadIdx.x;
  const int wave = tid >> 5;          // ni = 0..7
  const int lane = tid & 31;
  const int bb   = blockIdx.x;        // batch half: rows bb*16..bb*16+15
  const int ncol = wave * 16 + (lane & 15);
  const int mrow = (lane < 16) ? 0 : 8;

  // zero initial hidden state (both buffers)
  for (int i = tid; i < 2 * 16 * HS; i += 256) {
    ((bf16_t*)h0s)[i] = (bf16_t)0.0f;
    ((bf16_t*)h1s)[i] = (bf16_t)0.0f;
  }

  // Register-resident recurrent weights: 12 x v16bf = 96 VGPRs/wave.
  v16bf whh0[4], wih1[4], whh1[4];
#pragma unroll
  for (int ks = 0; ks < 4; ++ks) {
    whh0[ks] = load_B_g(Whh0, RNN_H, wave * 16, lane, ks * 32);
    wih1[ks] = load_B_g(Wih1, RNN_H, wave * 16, lane, ks * 32);
    whh1[ks] = load_B_g(Whh1, RNN_H, wave * 16, lane, ks * 32);
  }
  const float b1 = bs1[ncol];

  // Prefetch xp for t=0 (accumulator-layout, contiguous 32B per lane).
  const int tile = bb * 8 + wave;
  const size_t lane_off = (size_t)lane * 8;
  const float* xpl = xp + ((size_t)tile * 32) * 8 + lane_off;
  v4f p0 = *(const v4f*)(xpl + 0);
  v4f p1 = *(const v4f*)(xpl + 4);

  __syncthreads();

  for (int t = 0; t < RNN_T; ++t) {
    const int cur = t & 1, nxt = cur ^ 1;

    v8f acc0 = {p0[0], p0[1], p0[2], p0[3], p1[0], p1[1], p1[2], p1[3]};
    v8f acc1 = {b1, b1, b1, b1, b1, b1, b1, b1};

    if (t + 1 < RNN_T) {  // register prefetch of next step's xp
      const float* q = xp + (((size_t)(t + 1) * 16 + tile) * 32) * 8 + lane_off;
      p0 = *(const v4f*)(q + 0);
      p1 = *(const v4f*)(q + 4);
    }
    if (t + 8 < RNN_T) {  // L2 prefetch (global_prefetch_b8) 8 steps ahead
      __builtin_prefetch(xp + (((size_t)(t + 8) * 16 + tile) * 32) * 8 + lane_off,
                         0, 3);
    }

    // ---- pre-barrier work: two independent WMMA chains ----
    //   acc0 += h0(t-1) @ Whh0^T        (layer-0 recurrence)
    //   acc1 += h1(t-1) @ Whh1^T        (layer-1 recurrence, hoisted)
#pragma unroll
    for (int ks = 0; ks < 4; ++ks) {
      v16bf a0 = load_A_lds(&h0s[cur][0][0], lane, ks * 32, HS);
      v16bf a1 = load_A_lds(&h1s[cur][0][0], lane, ks * 32, HS);
      acc0 = WMMA_BF16(a0, whh0[ks], acc0);
      acc1 = WMMA_BF16(a1, whh1[ks], acc1);
    }
    {
      bf16_t* h0w = &h0s[nxt][0][0];
#pragma unroll
      for (int r = 0; r < 8; ++r) {
        float hv = fast_tanh(acc0[r]);
        h0w[(mrow + r) * HS + ncol] = (bf16_t)hv;
      }
    }
    __syncthreads();  // h0n visible to all waves

    // ---- post-barrier: only the h0n @ Wih1^T chain + tanh ----
#pragma unroll
    for (int ks = 0; ks < 4; ++ks) {
      v16bf a = load_A_lds(&h0s[nxt][0][0], lane, ks * 32, HS);
      acc1 = WMMA_BF16(a, wih1[ks], acc1);
    }
    {
      bf16_t* h1w = &h1s[nxt][0][0];
#pragma unroll
      for (int r = 0; r < 8; ++r) {
        float yv = fast_tanh(acc1[r]);
        int b = bb * 16 + mrow + r;
        out[((size_t)b * RNN_T + t) * RNN_H + ncol] = yv;  // coalesced 64B rows
        h1w[(mrow + r) * HS + ncol] = (bf16_t)yv;
      }
    }
    __syncthreads();  // h1n visible before next step reads it
  }
}

// ---------------------------------------------------------------------------
extern "C" void kernel_launch(void* const* d_in, const int* in_sizes, int n_in,
                              void* d_out, int out_size, void* d_ws, size_t ws_size,
                              hipStream_t stream) {
  (void)in_sizes; (void)n_in; (void)out_size; (void)ws_size;
  const float* x     = (const float*)d_in[0];
  const float* Wih0f = (const float*)d_in[1];
  const float* Whh0f = (const float*)d_in[2];
  const float* bih0  = (const float*)d_in[3];
  const float* bhh0  = (const float*)d_in[4];
  const float* Wih1f = (const float*)d_in[5];
  const float* Whh1f = (const float*)d_in[6];
  const float* bih1  = (const float*)d_in[7];
  const float* bhh1  = (const float*)d_in[8];
  float* out = (float*)d_out;

  char* ws = (char*)d_ws;
  float*  xp    = (float*)(ws + XP_OFF);
  bf16_t* wih0b = (bf16_t*)(ws + WIH0_OFF);
  bf16_t* whh0b = (bf16_t*)(ws + WHH0_OFF);
  bf16_t* wih1b = (bf16_t*)(ws + WIH1_OFF);
  bf16_t* whh1b = (bf16_t*)(ws + WHH1_OFF);
  float*  bs0   = (float*)(ws + BS0_OFF);
  float*  bs1   = (float*)(ws + BS1_OFF);

  rnn_prep<<<321, 256, 0, stream>>>(Wih0f, Whh0f, bih0, bhh0, Wih1f, Whh1f,
                                    bih1, bhh1, wih0b, whh0b, wih1b, whh1b,
                                    bs0, bs1);
  xp_gemm<<<RNN_T, 512, 0, stream>>>(x, wih0b, bs0, xp);
  rnn_scan<<<2, 256, 0, stream>>>(xp, whh0b, wih1b, whh1b, bs1, out);
}
